// PointNetSetAbstractionMsg_29635274342490
// MI455X (gfx1250) — matrix operand order repaired and checked by reading, not compiled, not run on hardware
//
#include <hip/hip_runtime.h>
#include <hip/hip_fp16.h>

#define BB 8
#define NN 4096
#define SS 1024
#define EPSF 1e-5f

typedef __attribute__((ext_vector_type(16))) _Float16 v16h;
typedef __attribute__((ext_vector_type(8)))  float    v8f;
typedef __attribute__((ext_vector_type(4)))  int      vi4;

union AV { v16h v; uint32_t u[8]; uint4 q[2]; };

#if defined(__has_builtin)
#if __has_builtin(__builtin_amdgcn_global_load_async_to_lds_b128) && \
    __has_builtin(__builtin_amdgcn_s_wait_asynccnt)
#define HAVE_ASYNC_LDS 1
#endif
#endif
#ifndef HAVE_ASYNC_LDS
#define HAVE_ASYNC_LDS 0
#endif

// ---------------- FPS: one block per batch, 16 points per thread in registers ----
__global__ __launch_bounds__(256) void fps_kernel(const float* __restrict__ xyz,
                                                  int* __restrict__ far) {
  const int b = blockIdx.x;
  const int t = threadIdx.x;
  const float* xb = xyz + (size_t)b * 3 * NN;
  constexpr int T = 256, P = NN / T; // 16
  float px[P], py[P], pz[P], d[P];
#pragma unroll
  for (int i = 0; i < P; ++i) {
    int n = i * T + t;
    px[i] = xb[n]; py[i] = xb[NN + n]; pz[i] = xb[2 * NN + n];
    d[i] = 1e10f;
  }
  __shared__ float sval[T];
  __shared__ int   sidx[T];
  int cur = 0;
  for (int it = 0; it < SS; ++it) {
    if (t == 0) far[b * SS + it] = cur;
    float cx = xb[cur], cy = xb[NN + cur], cz = xb[2 * NN + cur];
    float best = -1.0f; int bestn = 0x7fffffff;
#pragma unroll
    for (int i = 0; i < P; ++i) {
      float dx = px[i] - cx, dy = py[i] - cy, dz = pz[i] - cz;
      float dd = dx * dx + dy * dy + dz * dz;
      d[i] = fminf(d[i], dd);
      if (d[i] > best) { best = d[i]; bestn = i * T + t; }
    }
    sval[t] = best; sidx[t] = bestn;
    __syncthreads();
    for (int off = T / 2; off > 0; off >>= 1) {
      if (t < off) {
        float ov = sval[t + off]; int oi = sidx[t + off];
        if (ov > sval[t] || (ov == sval[t] && oi < sidx[t])) { sval[t] = ov; sidx[t] = oi; }
      }
      __syncthreads();
    }
    cur = sidx[0];
    __syncthreads();
  }
}

// ---------------- gather sampled centers: (B,S,3) scratch + (B,3,S) output ------
__global__ void gather_centers(const float* __restrict__ xyz, const int* __restrict__ far,
                               float* __restrict__ nxyz, float* __restrict__ out_xyz) {
  int t = blockIdx.x * blockDim.x + threadIdx.x;
  if (t >= BB * SS) return;
  int b = t / SS, s = t % SS;
  int n = far[t];
  const float* xb = xyz + (size_t)b * 3 * NN;
  float x = xb[n], y = xb[NN + n], z = xb[2 * NN + n];
  nxyz[t * 3 + 0] = x; nxyz[t * 3 + 1] = y; nxyz[t * 3 + 2] = z;
  out_xyz[(size_t)b * 3 * SS + 0 * SS + s] = x;
  out_xyz[(size_t)b * 3 * SS + 1 * SS + s] = y;
  out_xyz[(size_t)b * 3 * SS + 2 * SS + s] = z;
}

// ---------------- ball query: one wave32 per center, ballot-ordered first-K -----
__global__ __launch_bounds__(256) void ball_query(const float* __restrict__ xyz,
                                                  const float* __restrict__ nxyz,
                                                  int* __restrict__ idx, int K, float r2) {
  int gt = blockIdx.x * blockDim.x + threadIdx.x;
  int w = gt >> 5, lane = gt & 31;
  if (w >= BB * SS) return;
  int b = w / SS;
  const float* xb = xyz + (size_t)b * 3 * NN;
  float cx = nxyz[w * 3 + 0], cy = nxyz[w * 3 + 1], cz = nxyz[w * 3 + 2];
  int* out = idx + (size_t)w * K;
  int cnt = 0, first = NN - 1;
  for (int n0 = 0; n0 < NN && cnt < K; n0 += 32) {
    int n = n0 + lane;
    float dx = xb[n] - cx, dy = xb[NN + n] - cy, dz = xb[2 * NN + n] - cz;
    bool inb = (dx * dx + dy * dy + dz * dz <= r2);
    unsigned long long m = __ballot(inb);
    if (cnt == 0 && m) first = n0 + (int)__builtin_ctzll(m);
    int pos = cnt + (int)__popcll(m & ((1ull << lane) - 1ull));
    if (inb && pos < K) out[pos] = n;
    cnt += (int)__popcll(m);
  }
  int start = cnt < K ? cnt : K;
  for (int p = start + lane; p < K; p += 32) out[p] = first;
}

// ---------------- grouping: build f16 rows [pts(3) | rel_xyz(3) | zero pad..32] -
__global__ void group_kernel(const float* __restrict__ xyz, const float* __restrict__ pts,
                             const float* __restrict__ nxyz, const int* __restrict__ idx,
                             __half* __restrict__ X, int K) {
  size_t t = (size_t)blockIdx.x * blockDim.x + threadIdx.x;
  size_t total = (size_t)BB * SS * K;
  if (t >= total) return;
  size_t bs = t / K;
  int b = (int)(bs / SS);
  int n = idx[t];
  if (n < 0) n = 0;
  if (n > NN - 1) n = NN - 1;
  const float* xb = xyz + (size_t)b * 3 * NN;
  const float* pb = pts + (size_t)b * 3 * NN;
  float cx = nxyz[bs * 3 + 0], cy = nxyz[bs * 3 + 1], cz = nxyz[bs * 3 + 2];
  __half* row = X + t * 32;
  row[0] = (__half)pb[n];
  row[1] = (__half)pb[NN + n];
  row[2] = (__half)pb[2 * NN + n];
  row[3] = (__half)(xb[n] - cx);
  row[4] = (__half)(xb[NN + n] - cy);
  row[5] = (__half)(xb[2 * NN + n] - cz);
#pragma unroll
  for (int c = 6; c < 32; ++c) row[c] = (__half)0.0f;
}

// ---------------- weight prep: fp32 (Cout x CinA) -> f16 (Cout x CinP padded) ---
__global__ void prep_w(const float* __restrict__ W, __half* __restrict__ Wh,
                       int Cout, int CinA, int CinP) {
  int t = blockIdx.x * blockDim.x + threadIdx.x;
  if (t >= Cout * CinP) return;
  int c = t % CinP, o = t / CinP;
  Wh[t] = (c < CinA) ? (__half)W[o * CinA + c] : (__half)0.0f;
}

// ---------------- WMMA GEMM + fused bias + per-block BN partial stats -----------
// One wave owns one 16-row M tile and ALL Cout/16 N tiles (A fragment reused).
// Weights staged to LDS once per block (async-to-LDS when available).
// Per-block deterministic per-channel sum / sumsq written to Ps/Pq.
template <int CIN, int COUT>
__global__ __launch_bounds__(256) void gemm_wmma(const __half* __restrict__ X,
                                                 const __half* __restrict__ Wg,
                                                 const float* __restrict__ bias,
                                                 __half* __restrict__ Y,
                                                 float* __restrict__ Ps,
                                                 float* __restrict__ Pq) {
  constexpr int NCT = COUT / 16;
  __shared__ __align__(16) __half Wl[COUT * CIN];
  __shared__ float SlS[8 * COUT];
  __shared__ float SlQ[8 * COUT];

  // ---- stage weights into LDS (16B per thread-iteration) ----
  constexpr int CH16 = COUT * CIN / 8;
#if HAVE_ASYNC_LDS
  for (int i = threadIdx.x; i < CH16; i += 256) {
    __builtin_amdgcn_global_load_async_to_lds_b128(
        (__attribute__((address_space(1))) vi4*)(Wg + i * 8),
        (__attribute__((address_space(3))) vi4*)(Wl + i * 8), 0, 0);
  }
  __builtin_amdgcn_s_wait_asynccnt(0);
#else
  for (int i = threadIdx.x; i < CH16; i += 256)
    *(uint4*)(Wl + i * 8) = *(const uint4*)(Wg + i * 8);
#endif
  __syncthreads();

  const int lane = threadIdx.x & 31;
  const int wave = threadIdx.x >> 5;
  const int hi = lane >> 4;   // lanes 16..31
  const int mn = lane & 15;
  const size_t rowTile = (size_t)blockIdx.x * 8 + wave;
  const __half* xrow = X + (rowTile * 16 + mn) * (size_t)CIN;

  v8f acc[NCT];
#pragma unroll
  for (int ct = 0; ct < NCT; ++ct) acc[ct] = (v8f){};

#pragma unroll
  for (int kb = 0; kb < CIN; kb += 32) {
    AV a;
    // A 16x32 f16: VGPR v<4 -> K=2v(+8 hi); v>=4 -> K=8+2v(+8 hi); pairs contiguous.
    a.q[0] = *(const uint4*)(xrow + kb + (hi ? 8 : 0));
    a.q[1] = *(const uint4*)(xrow + kb + 16 + (hi ? 8 : 0));
#pragma unroll
    for (int ct = 0; ct < NCT; ++ct) {
      AV bm;
      // B 32x16 f16: lane holds column, K = 8h..8h+7 then 16+8h..16+8h+7 (same as A).
      const __half* wp = Wl + (size_t)(ct * 16 + mn) * CIN + kb + (hi ? 8 : 0);
      bm.q[0] = *(const uint4*)(wp);
      bm.q[1] = *(const uint4*)(wp + 16);
      acc[ct] = __builtin_amdgcn_wmma_f32_16x16x32_f16(false, a.v, false, bm.v,
                                                       (short)0, acc[ct], false, false);
    }
  }

  // ---- epilogue: bias, store f16, per-channel partial sums ----
  const size_t rbase = rowTile * 16 + (hi ? 8 : 0);
#pragma unroll
  for (int ct = 0; ct < NCT; ++ct) {
    const int col = ct * 16 + mn;
    const float bsv = bias[col];
    float s = 0.f, q = 0.f;
#pragma unroll
    for (int r = 0; r < 8; ++r) {
      float vv = acc[ct][r] + bsv;     // D: VGPR r -> row rbase+r, channel col
      s += vv; q += vv * vv;
      Y[(rbase + r) * (size_t)COUT + col] = (__half)vv;
    }
    s += __shfl_xor(s, 16, 32);        // combine row halves (lanes L, L+16: same col)
    q += __shfl_xor(q, 16, 32);
    if (lane < 16) { SlS[wave * COUT + col] = s; SlQ[wave * COUT + col] = q; }
  }
  __syncthreads();
  for (int c = threadIdx.x; c < COUT; c += 256) {
    float s = 0.f, q = 0.f;
#pragma unroll
    for (int w = 0; w < 8; ++w) { s += SlS[w * COUT + c]; q += SlQ[w * COUT + c]; }
    Ps[(size_t)blockIdx.x * COUT + c] = s;
    Pq[(size_t)blockIdx.x * COUT + c] = q;
  }
}

// ---------------- reduce per-block partials -> BN scale/shift -------------------
__global__ __launch_bounds__(256) void bn_finalize(const float* __restrict__ Ps,
                                                   const float* __restrict__ Pq,
                                                   const float* __restrict__ gamma,
                                                   const float* __restrict__ beta,
                                                   float* __restrict__ cA, float* __restrict__ cB,
                                                   int nb, int Cout, float invR) {
  int c = blockIdx.x, t = threadIdx.x;
  float s = 0.f, q = 0.f;
  for (int i = t; i < nb; i += 256) {
    s += Ps[(size_t)i * Cout + c];
    q += Pq[(size_t)i * Cout + c];
  }
  __shared__ float ss[256], sq[256];
  ss[t] = s; sq[t] = q;
  __syncthreads();
  for (int off = 128; off > 0; off >>= 1) {
    if (t < off) { ss[t] += ss[t + off]; sq[t] += sq[t + off]; }
    __syncthreads();
  }
  if (t == 0) {
    float mean = ss[0] * invR;
    float var = sq[0] * invR - mean * mean;
    float sc = gamma[c] * rsqrtf(var + EPSF);
    cA[c] = sc;
    cB[c] = beta[c] - mean * sc;
  }
}

// ---------------- normalize + ReLU in place -------------------------------------
__global__ void bn_relu(__half* __restrict__ Y, const float* __restrict__ cA,
                        const float* __restrict__ cB, size_t total, int Cout) {
  size_t t = (size_t)blockIdx.x * blockDim.x + threadIdx.x;
  if (t >= total) return;
  int c = (int)(t % Cout);
  float v = (float)Y[t];
  v = fmaxf(v * cA[c] + cB[c], 0.0f);
  Y[t] = (__half)v;
}

// ---------------- max over neighborhood K, write (B, 320, S) --------------------
__global__ void maxpool(const __half* __restrict__ X, float* __restrict__ outP,
                        int K, int Cout, int choff) {
  size_t t = (size_t)blockIdx.x * blockDim.x + threadIdx.x;
  size_t total = (size_t)BB * SS * Cout;
  if (t >= total) return;
  int c = (int)(t % Cout);
  size_t bs = t / Cout;
  int b = (int)(bs / SS), s = (int)(bs % SS);
  const __half* xp = X + bs * (size_t)K * Cout + c;
  float m = -1e30f;
  for (int k = 0; k < K; ++k) m = fmaxf(m, (float)xp[(size_t)k * Cout]);
  outP[(size_t)b * 320 * SS + (size_t)(choff + c) * SS + s] = m;
}

// ---------------- host-side dispatch --------------------------------------------
static void launch_gemm(int CinP, int Cout, const __half* X, const __half* Wg,
                        const float* bias, __half* Y, float* Ps, float* Pq,
                        size_t R, hipStream_t st) {
  unsigned nb = (unsigned)(R / 128);
  if (CinP == 32 && Cout == 32)
    gemm_wmma<32, 32><<<nb, 256, 0, st>>>(X, Wg, bias, Y, Ps, Pq);
  else if (CinP == 32 && Cout == 64)
    gemm_wmma<32, 64><<<nb, 256, 0, st>>>(X, Wg, bias, Y, Ps, Pq);
  else if (CinP == 64 && Cout == 64)
    gemm_wmma<64, 64><<<nb, 256, 0, st>>>(X, Wg, bias, Y, Ps, Pq);
  else if (CinP == 64 && Cout == 96)
    gemm_wmma<64, 96><<<nb, 256, 0, st>>>(X, Wg, bias, Y, Ps, Pq);
  else if (CinP == 64 && Cout == 128)
    gemm_wmma<64, 128><<<nb, 256, 0, st>>>(X, Wg, bias, Y, Ps, Pq);
  else if (CinP == 96 && Cout == 128)
    gemm_wmma<96, 128><<<nb, 256, 0, st>>>(X, Wg, bias, Y, Ps, Pq);
}

extern "C" void kernel_launch(void* const* d_in, const int* in_sizes, int n_in,
                              void* d_out, int out_size, void* d_ws, size_t ws_size,
                              hipStream_t stream) {
  (void)in_sizes; (void)n_in; (void)out_size; (void)ws_size;
  const float* xyz = (const float*)d_in[0];
  const float* pts = (const float*)d_in[1];
  float* out_xyz = (float*)d_out;
  float* out_pts = (float*)d_out + (size_t)BB * 3 * SS;

  char* ws = (char*)d_ws;
  size_t off = 0;
  auto carve = [&](size_t bytes) -> void* {
    void* p = ws + off;
    off = (off + bytes + 255) & ~(size_t)255;
    return p;
  };
  int*    far  = (int*)carve((size_t)BB * SS * 4);
  float*  nxyz = (float*)carve((size_t)BB * SS * 3 * 4);
  int*    idx  = (int*)carve((size_t)BB * SS * 128 * 4);
  float*  cA   = (float*)carve(128 * 4);
  float*  cB   = (float*)carve(128 * 4);
  __half* Wh   = (__half*)carve(128 * 128 * 2);
  float*  Ps   = (float*)carve((size_t)8192 * 128 * 4);   // per-block channel sums
  float*  Pq   = (float*)carve((size_t)8192 * 128 * 4);   // per-block channel sumsq
  size_t bufBytes = (size_t)BB * SS * 128 * 128 * 2;      // 256 MB each (max R x max C, f16)
  __half* bufA = (__half*)carve(bufBytes);
  __half* bufB = (__half*)carve(bufBytes);

  fps_kernel<<<BB, 256, 0, stream>>>(xyz, far);
  gather_centers<<<(BB * SS + 255) / 256, 256, 0, stream>>>(xyz, far, nxyz, out_xyz);

  const int   Ks[3]  = {16, 32, 128};
  const float rad[3] = {0.1f, 0.2f, 0.4f};
  const int   mlp[3][3] = {{32, 32, 64}, {64, 64, 128}, {64, 96, 128}};
  int choff = 0;
  for (int sc = 0; sc < 3; ++sc) {
    int K = Ks[sc];
    size_t R = (size_t)BB * SS * K;
    ball_query<<<(BB * SS * 32) / 256, 256, 0, stream>>>(xyz, nxyz, idx, K, rad[sc] * rad[sc]);
    group_kernel<<<(unsigned)((R + 255) / 256), 256, 0, stream>>>(xyz, pts, nxyz, idx, bufA, K);
    __half* cur = bufA;
    __half* nxt = bufB;
    int CinA = 6, CinP = 32;
    for (int j = 0; j < 3; ++j) {
      int Cout = mlp[sc][j];
      const float* W     = (const float*)d_in[2 + (sc * 3 + j) * 4 + 0];
      const float* bias  = (const float*)d_in[2 + (sc * 3 + j) * 4 + 1];
      const float* gamma = (const float*)d_in[2 + (sc * 3 + j) * 4 + 2];
      const float* beta  = (const float*)d_in[2 + (sc * 3 + j) * 4 + 3];
      prep_w<<<(Cout * CinP + 255) / 256, 256, 0, stream>>>(W, Wh, Cout, CinA, CinP);
      launch_gemm(CinP, Cout, cur, Wh, bias, nxt, Ps, Pq, R, stream);
      int nb = (int)(R / 128);
      bn_finalize<<<Cout, 256, 0, stream>>>(Ps, Pq, gamma, beta, cA, cB, nb, Cout,
                                            1.0f / (float)R);
      size_t total = R * (size_t)Cout;
      bn_relu<<<(unsigned)((total + 255) / 256), 256, 0, stream>>>(nxt, cA, cB, total, Cout);
      __half* tmp = cur; cur = nxt; nxt = tmp;
      CinA = Cout; CinP = Cout;
    }
    maxpool<<<(unsigned)(((size_t)BB * SS * mlp[sc][2] + 255) / 256), 256, 0, stream>>>(
        cur, out_pts, K, mlp[sc][2], choff);
    choff += mlp[sc][2];
  }
}
